// Convolution_12515534700915
// MI455X (gfx1250) — compile-verified, matrix operand order checked by reading, not run on hardware
//
#include <hip/hip_runtime.h>

typedef __attribute__((ext_vector_type(16))) _Float16 v16h;
typedef __attribute__((ext_vector_type(8)))  float    v8f;

namespace {
constexpr int kN  = 25000;
constexpr int kE  = 400000;
constexpr int kNB = 10;
constexpr int WPB = 8;   // waves per 256-thread block (wave32)
constexpr float kCS = 0.38268343236508978f;   // sin(pi/8)
constexpr float kCX = 0.92387953251128674f;   // cos(pi/8)
constexpr float kInvSqrt3  = 0.57735026918962584f;
constexpr float kInvSqrtNN = 0.25f;           // 1/sqrt(16)
}

// ---- CDNA5 16-bit A-layout swizzle ---------------------------------------
// Fragment read: lane l, element j (of 16) at k-tile kt corresponds to
//   m = l & 15, k = kt*32 + 8*(l>>4) + (j<8 ? j : 16 + (j-8)).
// Inverse (element (m,k) -> flat swizzled index):
__device__ __host__ __forceinline__ int a_swz_idx(int m, int k) {
  const int kt = k >> 5;
  const int kk = k & 31;
  const int half = (kk >> 3) & 1;
  const int j = (kk & 7) + (((kk >> 4) & 1) << 3);
  return ((kt * 32) + m + (half << 4)) * 16 + j;
}

// A fragment: one aligned 32B LDS load (2 x ds_load_b128)
__device__ __forceinline__ v16h frag_A_swz(const _Float16* base, int kt, int lane) {
  return *(const v16h*)(base + ((size_t)(kt * 32 + lane) << 4));
}

// B fragment from pre-swizzled global weight copy: one aligned 32B load
// layout: [kt][nt][lane 0..31][16 f16]
__device__ __forceinline__ v16h frag_B_swz(const _Float16* __restrict__ W, int ntn,
                                           int kt, int nt, int lane) {
  return *(const v16h*)(W + (((size_t)(kt * ntn + nt) * 32 + lane) << 4));
}

#define WMMA_F16(acc, a, b) \
  (acc) = __builtin_amdgcn_wmma_f32_16x16x32_f16(false, (a), false, (b), (short)0, (acc), false, false)

__device__ __forceinline__ void atomic_add_f32(float* p, float v) {
  __hip_atomic_fetch_add(p, v, __ATOMIC_RELAXED, __HIP_MEMORY_SCOPE_AGENT);
}

// ---- prep: f32 weights -> f16 copies pre-swizzled into B-fragment order --
// dst holds Kd*Nd f16 elements organized [kt][nt][lane][16]; zero padded.
__global__ void cvt_swz_f16_kernel(const float* __restrict__ src, _Float16* __restrict__ dst,
                                   int rs, int cs, int Kd, int Nd) {
  const int ntn = Nd >> 4;
  const int tot = Kd * Nd;
  for (int o = blockIdx.x * blockDim.x + threadIdx.x; o < tot; o += gridDim.x * blockDim.x) {
    const int tile = o >> 9;
    const int lane = (o >> 4) & 31;
    const int j = o & 15;
    const int kt = tile / ntn;
    const int nt = tile - kt * ntn;
    const int n = (nt << 4) + (lane & 15);
    const int k = (kt << 5) + ((lane >> 4) << 3) + ((j < 8) ? j : (8 + j));
    const float v = (k < rs && n < cs) ? src[(size_t)k * cs + n] : 0.0f;
    dst[o] = (_Float16)v;
  }
}

__global__ void zero_f32_kernel(float* __restrict__ p, size_t n) {
  for (size_t i = blockIdx.x * (size_t)blockDim.x + threadIdx.x; i < n;
       i += (size_t)gridDim.x * blockDim.x)
    p[i] = 0.0f;
}

// ---- kernel 1: node precompute -------------------------------------------
__global__ __launch_bounds__(256) void node_kernel(
    const float* __restrict__ xin, const float* __restrict__ attr_a,
    const float* __restrict__ attr_c,
    const _Float16* __restrict__ Wsa, const _Float16* __restrict__ Wsc,
    const _Float16* __restrict__ Wla, const _Float16* __restrict__ Wlc,
    const _Float16* __restrict__ Vsa, const _Float16* __restrict__ Vsc,
    const _Float16* __restrict__ Vla, const _Float16* __restrict__ Vlc,
    float* __restrict__ out, float* __restrict__ x1s, float* __restrict__ x1v) {
  __shared__ __align__(32) _Float16 As[WPB][1024];   // 16x64 swizzled (2 k-tiles)
  __shared__ float At[WPB][16][2];
  const int wave = threadIdx.x >> 5;
  const int lane = threadIdx.x & 31;
  const int mb = (lane >> 4) << 3;
  const int nl = lane & 15;
  const int ntiles = (kN + 15) / 16;
  for (int t0 = blockIdx.x * WPB; t0 < ntiles; t0 += gridDim.x * WPB) {
    const int tile = t0 + wave;
    const bool act = tile < ntiles;
    const int n0 = tile * 16;
    if (act) {
      for (int i = lane; i < 16 * 64; i += 32) {
        const int m = i >> 6, j = i & 63;
        const int node = n0 + m;
        As[wave][a_swz_idx(m, j)] =
            (_Float16)((node < kN) ? xin[(size_t)node * 160 + j] : 0.0f);
      }
      if (lane < 16) {
        const int node = n0 + lane;
        At[wave][lane][0] = (node < kN) ? attr_a[node] : 0.0f;
        At[wave][lane][1] = (node < kN) ? attr_c[node] : 0.0f;
      }
    }
    __syncthreads();
    if (act) {
#pragma unroll
      for (int ct = 0; ct < 4; ++ct) {
        v8f Pa = {}, Pc = {}, Ra = {}, Rc = {};
#pragma unroll
        for (int k = 0; k < 2; ++k) {
          const v16h a = frag_A_swz(As[wave], k, lane);
          WMMA_F16(Pa, a, frag_B_swz(Wsa, 4, k, ct, lane));
          WMMA_F16(Pc, a, frag_B_swz(Wsc, 4, k, ct, lane));
          WMMA_F16(Ra, a, frag_B_swz(Wla, 4, k, ct, lane));
          WMMA_F16(Rc, a, frag_B_swz(Wlc, 4, k, ct, lane));
        }
#pragma unroll
        for (int r = 0; r < 8; ++r) {
          const int m = mb + r;
          const int node = n0 + m;
          if (node < kN) {
            const float aA = At[wave][m][0], cA = At[wave][m][1];
            const int n = ct * 16 + nl;
            out[(size_t)node * 160 + n] = kCS * (aA * Pa[r] + cA * Pc[r]);
            x1s[(size_t)node * 64 + n]  = aA * Ra[r] + cA * Rc[r];
          }
        }
      }
    }
#pragma unroll 1
    for (int c = 0; c < 3; ++c) {
      __syncthreads();
      if (act) {
        for (int i = lane; i < 16 * 32; i += 32) {
          const int m = i >> 5, u = i & 31;
          const int node = n0 + m;
          As[wave][a_swz_idx(m, u)] =
              (_Float16)((node < kN) ? xin[(size_t)node * 160 + 64 + u * 3 + c] : 0.0f);
        }
      }
      __syncthreads();
      if (act) {
        const v16h a = frag_A_swz(As[wave], 0, lane);
#pragma unroll
        for (int ct = 0; ct < 2; ++ct) {
          v8f Pa = {}, Pc = {}, Ra = {}, Rc = {};
          WMMA_F16(Pa, a, frag_B_swz(Vsa, 2, 0, ct, lane));
          WMMA_F16(Pc, a, frag_B_swz(Vsc, 2, 0, ct, lane));
          WMMA_F16(Ra, a, frag_B_swz(Vla, 2, 0, ct, lane));
          WMMA_F16(Rc, a, frag_B_swz(Vlc, 2, 0, ct, lane));
#pragma unroll
          for (int r = 0; r < 8; ++r) {
            const int m = mb + r;
            const int node = n0 + m;
            if (node < kN) {
              const float aA = At[wave][m][0], cA = At[wave][m][1];
              const int u = ct * 16 + nl;
              out[(size_t)node * 160 + 64 + u * 3 + c] = kCS * (aA * Pa[r] + cA * Pc[r]);
              x1v[(size_t)node * 96 + u * 3 + c]       = aA * Ra[r] + cA * Rc[r];
            }
          }
        }
      }
    }
    __syncthreads();
  }
}

// ---- kernel 2: fused edge MLP (WMMA) + tensor product + atomic scatter ---
__global__ __launch_bounds__(256) void edge_kernel(
    const float* __restrict__ elen, const float* __restrict__ eattr,
    const int* __restrict__ esrc, const int* __restrict__ edst,
    const _Float16* __restrict__ Wfc1, const _Float16* __restrict__ Wfc2,
    const float* __restrict__ x1s, const float* __restrict__ x1v,
    float* __restrict__ agg) {
  __shared__ __align__(32) _Float16 Ast[WPB][512];   // 16x32 swizzled
  __shared__ __align__(32) _Float16 Hst[WPB][2048];  // 16x128 swizzled (4 k-tiles)
  __shared__ int   Ssrc[WPB][16];
  __shared__ int   Sdst[WPB][16];
  __shared__ float Ses[WPB][16];
  __shared__ float Sev[WPB][16][3];
  const int wave = threadIdx.x >> 5;
  const int lane = threadIdx.x & 31;
  const int mb = (lane >> 4) << 3;
  const int nl = lane & 15;
  const int ntiles = kE / 16;
  for (int t0 = blockIdx.x * WPB; t0 < ntiles; t0 += gridDim.x * WPB) {
    const int tile = t0 + wave;
    const bool act = tile < ntiles;
    const int e0 = tile * 16;
    if (act) {
      for (int i = lane; i < 16 * 32; i += 32) {
        const int m = i >> 5, k = i & 31;
        Ast[wave][a_swz_idx(m, k)] =
            (_Float16)((k < kNB) ? elen[(size_t)(e0 + m) * kNB + k] : 0.0f);
      }
      if (lane < 16) {
        const int e = e0 + lane;
        Ssrc[wave][lane] = esrc[e];
        Sdst[wave][lane] = edst[e];
        Ses[wave][lane] = eattr[(size_t)e * 4 + 0];
        Sev[wave][lane][0] = eattr[(size_t)e * 4 + 1];
        Sev[wave][lane][1] = eattr[(size_t)e * 4 + 2];
        Sev[wave][lane][2] = eattr[(size_t)e * 4 + 3];
      }
    }
    __syncthreads();
    if (act) {
      // FC1 (K padded 10->32) + silu, re-stage to LDS in swizzled A layout
      const v16h a1 = frag_A_swz(Ast[wave], 0, lane);
#pragma unroll
      for (int ct = 0; ct < 7; ++ct) {
        v8f h = {};
        WMMA_F16(h, a1, frag_B_swz(Wfc1, 7, 0, ct, lane));
#pragma unroll
        for (int r = 0; r < 8; ++r) {
          const float x = h[r];
          Hst[wave][a_swz_idx(mb + r, ct * 16 + nl)] =
              (_Float16)(x / (1.0f + __expf(-x)));
        }
      }
#pragma unroll
      for (int r = 0; r < 8; ++r)
        Hst[wave][a_swz_idx(mb + r, 112 + nl)] = (_Float16)0.0f;
    }
    __syncthreads();
    if (act) {
      // FC2: 12 col tiles x 4 k-steps (hidden padded 100->128)
      v8f acc[12] = {};
#pragma unroll
      for (int k = 0; k < 4; ++k) {
        const v16h a = frag_A_swz(Hst[wave], k, lane);
#pragma unroll
        for (int ct = 0; ct < 12; ++ct)
          WMMA_F16(acc[ct], a, frag_B_swz(Wfc2, 12, k, ct, lane));
      }
      int sm[8], dm[8];
      float es[8], ev0[8], ev1[8], ev2[8];
#pragma unroll
      for (int r = 0; r < 8; ++r) {
        const int m = mb + r;
        sm[r] = Ssrc[wave][m];  dm[r] = Sdst[wave][m];
        es[r] = Ses[wave][m];
        ev0[r] = Sev[wave][m][0]; ev1[r] = Sev[wave][m][1]; ev2[r] = Sev[wave][m][2];
      }
      // w0 (cols 0..63): mid0a -> agg[:, n]
#pragma unroll
      for (int ct = 0; ct < 4; ++ct) {
        const int n = ct * 16 + nl;
#pragma unroll
        for (int r = 0; r < 8; ++r) {
          const float w = acc[ct][r] * kInvSqrtNN;
          const float g = x1s[(size_t)sm[r] * 64 + n];
          atomic_add_f32(&agg[(size_t)dm[r] * 384 + n], w * g * es[r]);
        }
      }
      // w1 (cols 64..127): mid1a -> agg[:, 96+3u+c]
#pragma unroll
      for (int ct = 4; ct < 8; ++ct) {
        const int u = ct * 16 + nl - 64;
#pragma unroll
        for (int r = 0; r < 8; ++r) {
          const float w = acc[ct][r] * kInvSqrtNN;
          const float g = x1s[(size_t)sm[r] * 64 + u];
          float* base = &agg[(size_t)dm[r] * 384 + 96 + u * 3];
          atomic_add_f32(base + 0, w * g * ev0[r]);
          atomic_add_f32(base + 1, w * g * ev1[r]);
          atomic_add_f32(base + 2, w * g * ev2[r]);
        }
      }
      // w2 (cols 128..159): mid1b -> agg[:, 288+3u+c]
#pragma unroll
      for (int ct = 8; ct < 10; ++ct) {
        const int u = ct * 16 + nl - 128;
#pragma unroll
        for (int r = 0; r < 8; ++r) {
          const float w = acc[ct][r] * kInvSqrtNN * es[r];
          const float* gv = &x1v[(size_t)sm[r] * 96 + u * 3];
          float* base = &agg[(size_t)dm[r] * 384 + 288 + u * 3];
          atomic_add_f32(base + 0, w * gv[0]);
          atomic_add_f32(base + 1, w * gv[1]);
          atomic_add_f32(base + 2, w * gv[2]);
        }
      }
      // w3 (cols 160..191): mid0b -> agg[:, 64+u]
#pragma unroll
      for (int ct = 10; ct < 12; ++ct) {
        const int u = ct * 16 + nl - 160;
#pragma unroll
        for (int r = 0; r < 8; ++r) {
          const float w = acc[ct][r] * kInvSqrtNN * kInvSqrt3;
          const float* gv = &x1v[(size_t)sm[r] * 96 + u * 3];
          const float dot = gv[0] * ev0[r] + gv[1] * ev1[r] + gv[2] * ev2[r];
          atomic_add_f32(&agg[(size_t)dm[r] * 384 + 64 + u], w * dot);
        }
      }
    }
    __syncthreads();
  }
}

// ---- kernel 3: l2 mixes + l3 + final combine into d_out ------------------
__global__ __launch_bounds__(256) void out_kernel(
    const float* __restrict__ agg, const float* __restrict__ attr_a,
    const float* __restrict__ attr_c, const float* __restrict__ sym,
    const _Float16* __restrict__ L2sa, const _Float16* __restrict__ L2sc,
    const _Float16* __restrict__ L2va, const _Float16* __restrict__ L2vc,
    const _Float16* __restrict__ L3s, const _Float16* __restrict__ L3v,
    float* __restrict__ out) {
  __shared__ __align__(32) _Float16 A1[WPB][1536];   // 16x96 swizzled (3 k-tiles)
  __shared__ __align__(32) _Float16 A2[WPB][1536];
  __shared__ float At[WPB][16][3];
  const int wave = threadIdx.x >> 5;
  const int lane = threadIdx.x & 31;
  const int mb = (lane >> 4) << 3;
  const int nl = lane & 15;
  const int ntiles = (kN + 15) / 16;
  for (int t0 = blockIdx.x * WPB; t0 < ntiles; t0 += gridDim.x * WPB) {
    const int tile = t0 + wave;
    const bool act = tile < ntiles;
    const int n0 = tile * 16;
    if (act) {
      for (int i = lane; i < 16 * 96; i += 32) {
        const int m = i / 96, u = i % 96;
        const int node = n0 + m;
        A1[wave][a_swz_idx(m, u)] =
            (_Float16)((node < kN) ? agg[(size_t)node * 384 + u] : 0.0f);
      }
      if (lane < 16) {
        const int node = n0 + lane;
        At[wave][lane][0] = (node < kN) ? attr_a[node] : 0.0f;
        At[wave][lane][1] = (node < kN) ? attr_c[node] : 0.0f;
        At[wave][lane][2] = (node < kN) ? sym[node] : 0.0f;
      }
    }
    __syncthreads();
    if (act) {
#pragma unroll
      for (int ct = 0; ct < 6; ++ct) {
        v8f P = {}, Q = {};
#pragma unroll
        for (int k = 0; k < 3; ++k) {
          const v16h a = frag_A_swz(A1[wave], k, lane);
          WMMA_F16(P, a, frag_B_swz(L2sa, 6, k, ct, lane));
          WMMA_F16(Q, a, frag_B_swz(L2sc, 6, k, ct, lane));
        }
#pragma unroll
        for (int r = 0; r < 8; ++r) {
          const int m = mb + r;
          const float t = At[wave][m][0] * P[r] + At[wave][m][1] * Q[r];
          A2[wave][a_swz_idx(m, ct * 16 + nl)] = (_Float16)(At[wave][m][2] * t);
        }
      }
    }
    __syncthreads();
    if (act) {
#pragma unroll
      for (int ct = 0; ct < 4; ++ct) {
        v8f O = {};
#pragma unroll
        for (int k = 0; k < 3; ++k) {
          const v16h a = frag_A_swz(A2[wave], k, lane);
          WMMA_F16(O, a, frag_B_swz(L3s, 4, k, ct, lane));
        }
#pragma unroll
        for (int r = 0; r < 8; ++r) {
          const int node = n0 + mb + r;
          if (node < kN) out[(size_t)node * 160 + ct * 16 + nl] += kCX * O[r];
        }
      }
    }
#pragma unroll 1
    for (int c = 0; c < 3; ++c) {
      __syncthreads();
      if (act) {
        for (int i = lane; i < 16 * 96; i += 32) {
          const int m = i / 96, u = i % 96;
          const int node = n0 + m;
          A1[wave][a_swz_idx(m, u)] =
              (_Float16)((node < kN) ? agg[(size_t)node * 384 + 96 + u * 3 + c] : 0.0f);
        }
      }
      __syncthreads();
      if (act) {
#pragma unroll
        for (int ct = 0; ct < 6; ++ct) {
          v8f P = {}, Q = {};
#pragma unroll
          for (int k = 0; k < 3; ++k) {
            const v16h a = frag_A_swz(A1[wave], k, lane);
            WMMA_F16(P, a, frag_B_swz(L2va, 6, k, ct, lane));
            WMMA_F16(Q, a, frag_B_swz(L2vc, 6, k, ct, lane));
          }
#pragma unroll
          for (int r = 0; r < 8; ++r) {
            const int m = mb + r;
            const float t = At[wave][m][0] * P[r] + At[wave][m][1] * Q[r];
            A2[wave][a_swz_idx(m, ct * 16 + nl)] = (_Float16)(At[wave][m][2] * t);
          }
        }
      }
      __syncthreads();
      if (act) {
#pragma unroll
        for (int ct = 0; ct < 2; ++ct) {
          v8f O = {};
#pragma unroll
          for (int k = 0; k < 3; ++k) {
            const v16h a = frag_A_swz(A2[wave], k, lane);
            WMMA_F16(O, a, frag_B_swz(L3v, 2, k, ct, lane));
          }
#pragma unroll
          for (int r = 0; r < 8; ++r) {
            const int node = n0 + mb + r;
            if (node < kN) {
              const int u = ct * 16 + nl;
              out[(size_t)node * 160 + 64 + u * 3 + c] += kCX * O[r];
            }
          }
        }
      }
    }
    __syncthreads();
  }
}

extern "C" void kernel_launch(void* const* d_in, const int* in_sizes, int n_in,
                              void* d_out, int out_size, void* d_ws, size_t ws_size,
                              hipStream_t stream) {
  (void)in_sizes; (void)n_in; (void)out_size; (void)ws_size;
  const float* node_input = (const float*)d_in[0];
  const float* node_attr  = (const float*)d_in[1];
  const float* node_cattr = (const float*)d_in[2];
  const float* sym_mask   = (const float*)d_in[3];
  const float* edge_attr  = (const float*)d_in[4];
  const float* elen       = (const float*)d_in[5];
  const int*   esrc       = (const int*)d_in[6];
  const int*   edst       = (const int*)d_in[7];
  const float* Ws_sc_a = (const float*)d_in[8];
  const float* Wv_sc_a = (const float*)d_in[9];
  const float* Ws_sc_c = (const float*)d_in[10];
  const float* Wv_sc_c = (const float*)d_in[11];
  const float* Ws_l1_a = (const float*)d_in[12];
  const float* Wv_l1_a = (const float*)d_in[13];
  const float* Ws_l1_c = (const float*)d_in[14];
  const float* Wv_l1_c = (const float*)d_in[15];
  const float* W_fc1   = (const float*)d_in[16];
  const float* W_fc2   = (const float*)d_in[17];
  const float* Ws_l2_a = (const float*)d_in[18];
  const float* Wv_l2_a = (const float*)d_in[19];
  const float* Ws_l2_c = (const float*)d_in[20];
  const float* Wv_l2_c = (const float*)d_in[21];
  const float* Ws_l3   = (const float*)d_in[22];
  const float* Wv_l3   = (const float*)d_in[23];
  float* out = (float*)d_out;

  // workspace: agg[N,384] | x1s[N,64] | x1v[N,96] | swizzled f16 weights
  float* agg = (float*)d_ws;
  float* x1s = agg + (size_t)kN * 384;
  float* x1v = x1s + (size_t)kN * 64;
  _Float16* fh = (_Float16*)(x1v + (size_t)kN * 96);
  _Float16* hFc1 = fh;  fh += 32 * 112;   // K 10->32, Nout 100->112
  _Float16* hFc2 = fh;  fh += 128 * 192;  // K 100->128
  _Float16* hWsa = fh;  fh += 64 * 64;
  _Float16* hWsc = fh;  fh += 64 * 64;
  _Float16* hWla = fh;  fh += 64 * 64;
  _Float16* hWlc = fh;  fh += 64 * 64;
  _Float16* hVsa = fh;  fh += 32 * 32;
  _Float16* hVsc = fh;  fh += 32 * 32;
  _Float16* hVla = fh;  fh += 32 * 32;
  _Float16* hVlc = fh;  fh += 32 * 32;
  _Float16* hL2sa = fh; fh += 96 * 96;
  _Float16* hL2sc = fh; fh += 96 * 96;
  _Float16* hL2va = fh; fh += 96 * 96;
  _Float16* hL2vc = fh; fh += 96 * 96;
  _Float16* hL3s = fh;  fh += 96 * 64;
  _Float16* hL3v = fh;  fh += 96 * 32;

  cvt_swz_f16_kernel<<<32, 256, 0, stream>>>(W_fc1, hFc1, 10, 100, 32, 112);
  cvt_swz_f16_kernel<<<32, 256, 0, stream>>>(W_fc2, hFc2, 100, 192, 128, 192);
  cvt_swz_f16_kernel<<<32, 256, 0, stream>>>(Ws_sc_a, hWsa, 64, 64, 64, 64);
  cvt_swz_f16_kernel<<<32, 256, 0, stream>>>(Ws_sc_c, hWsc, 64, 64, 64, 64);
  cvt_swz_f16_kernel<<<32, 256, 0, stream>>>(Ws_l1_a, hWla, 64, 64, 64, 64);
  cvt_swz_f16_kernel<<<32, 256, 0, stream>>>(Ws_l1_c, hWlc, 64, 64, 64, 64);
  cvt_swz_f16_kernel<<<32, 256, 0, stream>>>(Wv_sc_a, hVsa, 32, 32, 32, 32);
  cvt_swz_f16_kernel<<<32, 256, 0, stream>>>(Wv_sc_c, hVsc, 32, 32, 32, 32);
  cvt_swz_f16_kernel<<<32, 256, 0, stream>>>(Wv_l1_a, hVla, 32, 32, 32, 32);
  cvt_swz_f16_kernel<<<32, 256, 0, stream>>>(Wv_l1_c, hVlc, 32, 32, 32, 32);
  cvt_swz_f16_kernel<<<32, 256, 0, stream>>>(Ws_l2_a, hL2sa, 96, 96, 96, 96);
  cvt_swz_f16_kernel<<<32, 256, 0, stream>>>(Ws_l2_c, hL2sc, 96, 96, 96, 96);
  cvt_swz_f16_kernel<<<32, 256, 0, stream>>>(Wv_l2_a, hL2va, 96, 96, 96, 96);
  cvt_swz_f16_kernel<<<32, 256, 0, stream>>>(Wv_l2_c, hL2vc, 96, 96, 96, 96);
  cvt_swz_f16_kernel<<<32, 256, 0, stream>>>(Ws_l3, hL3s, 96, 64, 96, 64);
  cvt_swz_f16_kernel<<<32, 256, 0, stream>>>(Wv_l3, hL3v, 96, 32, 96, 32);

  zero_f32_kernel<<<2048, 256, 0, stream>>>(agg, (size_t)kN * 384);

  node_kernel<<<200, 256, 0, stream>>>(node_input, node_attr, node_cattr,
                                       hWsa, hWsc, hWla, hWlc,
                                       hVsa, hVsc, hVla, hVlc,
                                       out, x1s, x1v);
  edge_kernel<<<1250, 256, 0, stream>>>(elen, edge_attr, esrc, edst,
                                        hFc1, hFc2, x1s, x1v, agg);
  out_kernel<<<200, 256, 0, stream>>>(agg, node_attr, node_cattr, sym_mask,
                                      hL2sa, hL2sc, hL2va, hL2vc,
                                      hL3s, hL3v, out);
}
